// StarAggregator_57878979281135
// MI455X (gfx1250) — matrix operand order repaired and checked by reading, not compiled, not run on hardware
//
#include <hip/hip_runtime.h>
#include <math.h>

// ---------------------------------------------------------------------------
// StarAggregator for MI455X (gfx1250), wave32 + V_WMMA_F32_16X16X32_BF16.
// One workgroup (8 wave32) per batch element; full 64x128 state in LDS.
// Rank-1 folding removes the sate@q_1 / sate@k_2 N*D*D matmuls entirely.
// ---------------------------------------------------------------------------

typedef __bf16 bf16_t;
typedef __attribute__((ext_vector_type(8)))  __bf16 v8bf;
typedef __attribute__((ext_vector_type(16))) __bf16 v16bf;
typedef __attribute__((ext_vector_type(8)))  float  v8f;

#define DEV static __device__ __forceinline__

DEV v16bf bf_cat(v8bf lo, v8bf hi) {
  return __builtin_shufflevector(lo, hi, 0,1,2,3,4,5,6,7,8,9,10,11,12,13,14,15);
}

// A fragment (16x32 bf16) from row-major buf[row][K], row stride S elements.
// ISA layout: lane(m=lane&15,h=lane>>4), elem e -> A[m][ (e>>3)*16 + h*8 + (e&7) + kb ]
// -> two contiguous 16B loads per lane.
DEV v16bf load_a_frag(const bf16_t* buf, int rowBase, int S, int kb, int lane) {
  const int m = lane & 15, h = lane >> 4;
  const bf16_t* p = buf + (size_t)(rowBase + m) * S + kb + h * 8;
  v8bf lo = *(const v8bf*)p;
  v8bf hi = *(const v8bf*)(p + 16);
  return bf_cat(lo, hi);
}

// B fragment (32x16 bf16) gathered from TRANSPOSED storage bt[N][K], stride S.
// ISA layout: lane(n=lane&15,h=lane>>4), elem e -> B[K = kb + (e>>3)*16 + h*8 + (e&7)][n]
// = bt[n][kb + (e>>3)*16 + h*8 + (e&7)] -> two contiguous 16B loads per lane.
DEV v16bf load_bt_frag(const bf16_t* buf, int nBase, int S, int kb, int lane) {
  const int n = lane & 15, h = lane >> 4;
  const bf16_t* p = buf + (size_t)(nBase + n) * S + kb + h * 8;
  v8bf lo = *(const v8bf*)p;
  v8bf hi = *(const v8bf*)(p + 16);
  return bf_cat(lo, hi);
}

DEV v16bf bf_mul(v16bf a, v16bf b) {
  v16bf r;
  #pragma unroll
  for (int i = 0; i < 16; ++i) r[i] = (bf16_t)((float)a[i] * (float)b[i]);
  return r;
}

DEV v8f wmma_bf16(v16bf a, v16bf b, v8f c) {
  // (neg_a, A, neg_b, B, c_mod, C, reuse_a, reuse_b)
  return __builtin_amdgcn_wmma_f32_16x16x32_bf16(false, a, false, b, (short)0, c,
                                                 false, false);
}

DEV float wave_max(float x) {
  #pragma unroll
  for (int o = 16; o > 0; o >>= 1) x = fmaxf(x, __shfl_xor(x, o, 32));
  return x;
}
DEV float wave_sum(float x) {
  #pragma unroll
  for (int o = 16; o > 0; o >>= 1) x += __shfl_xor(x, o, 32);
  return x;
}

__global__ void wlin_to_bf16(const float* __restrict__ w, bf16_t* __restrict__ o, int n) {
  int i = blockIdx.x * blockDim.x + threadIdx.x;
  if (i < n) o[i] = (bf16_t)w[i];
}

__global__ __launch_bounds__(256)
void star_agg_kernel(const float* __restrict__ hidden, const int* __restrict__ adj,
                     const float* __restrict__ mask,
                     const float* __restrict__ a0, const float* __restrict__ a1,
                     const float* __restrict__ a2, const float* __restrict__ a3,
                     const float* __restrict__ q1, const float* __restrict__ k1,
                     const float* __restrict__ q2, const float* __restrict__ k2,
                     const bf16_t* __restrict__ wl,   // bf16 W_lin [128][256]
                     float* __restrict__ out, float* __restrict__ star_out) {
  constexpr int Nn = 64, Dd = 128;
  constexpr float INV_SQRT_D = 0.08838834764831845f;   // 1/sqrt(128)

  __shared__ alignas(16) float  sS [Nn * Dd];   // f32 master sate
  __shared__ alignas(16) bf16_t hsS[Nn * Dd];   // bf16 sate, row-major [n][d]
  __shared__ alignas(16) bf16_t htS[Dd * Nn];   // bf16 sate, transposed [d][n]
  __shared__ alignas(16) bf16_t hbS[Nn * Dd];   // bf16 original hidden
  __shared__ alignas(16) float  lgS[Nn * Nn];   // attention logits (f32)
  __shared__ alignas(16) bf16_t atS[Nn * Nn];   // softmaxed attention (bf16)
  __shared__ alignas(16) bf16_t abS[4 * Dd];    // a_0..a_3 (bf16)
  __shared__ unsigned char      adS[Nn * Nn];   // adjacency labels 0..4
  __shared__ float stS[Dd], kvS[Dd], qvS[Dd], wS[Dd], uS[Dd];
  __shared__ float alS[Nn], beS[Nn], mkS[Nn];

  const int tid  = threadIdx.x;
  const int lane = tid & 31;
  const int wid  = tid >> 5;
  const int b    = blockIdx.x;

  const float* Hg = hidden + (size_t)b * Nn * Dd;
  const int*   Ag = adj    + (size_t)b * Nn * Nn;

  // ---- load batch state into LDS ----
  for (int idx = tid; idx < Nn * Dd; idx += 256) {
    float v = Hg[idx];
    int n = idx >> 7, d = idx & 127;
    sS[idx] = v;
    bf16_t bv = (bf16_t)v;
    hsS[idx] = bv; hbS[idx] = bv; htS[d * Nn + n] = bv;
  }
  for (int idx = tid; idx < Nn * Nn; idx += 256)
    adS[idx] = (unsigned char)Ag[idx];
  if (tid < Nn) mkS[tid] = mask[(size_t)b * Nn + tid];
  if (tid < Dd) {
    abS[0 * Dd + tid] = (bf16_t)a0[tid];
    abS[1 * Dd + tid] = (bf16_t)a1[tid];
    abS[2 * Dd + tid] = (bf16_t)a2[tid];
    abS[3 * Dd + tid] = (bf16_t)a3[tid];
  }
  __syncthreads();

  // star = masked average of hidden rows
  if (tid < Dd) {
    float s = 0.f, ms = 0.f;
    for (int n = 0; n < Nn; ++n) { s += sS[n * Dd + tid] * mkS[n]; ms += mkS[n]; }
    stS[tid] = s / ms;
  }
  __syncthreads();

  for (int step = 0; step < 2; ++step) {
    // ---- e_k = leaky((h*a_k) h^T), select by adj -> logits (WMMA) ----
    for (int t = wid; t < 16; t += 8) {
      const int ti = t >> 2, tj = t & 3;
      v8f acc0 = {0,0,0,0,0,0,0,0}, acc1 = acc0, acc2 = acc0, acc3 = acc0;
      #pragma unroll
      for (int kk = 0; kk < 4; ++kk) {
        const int kb = kk * 32;
        v16bf hf  = load_a_frag(hsS, ti * 16, Dd, kb, lane);
        v16bf bfb = load_bt_frag(hsS, tj * 16, Dd, kb, lane);  // B^T == h rows
        acc0 = wmma_bf16(bf_mul(hf, load_a_frag(abS + 0 * Dd, 0, 0, kb, lane)), bfb, acc0);
        acc1 = wmma_bf16(bf_mul(hf, load_a_frag(abS + 1 * Dd, 0, 0, kb, lane)), bfb, acc1);
        acc2 = wmma_bf16(bf_mul(hf, load_a_frag(abS + 2 * Dd, 0, 0, kb, lane)), bfb, acc2);
        acc3 = wmma_bf16(bf_mul(hf, load_a_frag(abS + 3 * Dd, 0, 0, kb, lane)), bfb, acc3);
      }
      const int hh = lane >> 4, j = tj * 16 + (lane & 15);
      #pragma unroll
      for (int r = 0; r < 8; ++r) {
        const int i  = ti * 16 + hh * 8 + r;
        const int a8 = adS[i * Nn + j];
        float x  = (a8 == 1) ? acc0[r] : (a8 == 2) ? acc1[r]
                 : (a8 == 3) ? acc2[r] : acc3[r];
        float el = (x >= 0.f) ? x : 0.2f * x;
        lgS[i * Nn + j] = (a8 >= 1 && a8 <= 4) ? el : -9.0e15f;
      }
    }
    __syncthreads();

    // ---- row softmax (one wave per 8 rows, wave32 shuffles) ----
    for (int rr = 0; rr < 8; ++rr) {
      const int i = wid * 8 + rr;
      float l0 = lgS[i * Nn + lane], l1 = lgS[i * Nn + 32 + lane];
      float m  = wave_max(fmaxf(l0, l1));
      float e0 = __expf(l0 - m), e1 = __expf(l1 - m);
      float inv = 1.0f / wave_sum(e0 + e1);
      atS[i * Nn + lane]      = (bf16_t)(e0 * inv);
      atS[i * Nn + 32 + lane] = (bf16_t)(e1 * inv);
    }
    __syncthreads();

    // ---- sate = att @ h (WMMA, K=64) -> f32 master ----
    for (int t = wid; t < 32; t += 8) {
      const int ti = t >> 3, td = t & 7;
      v8f acc = {0,0,0,0,0,0,0,0};
      #pragma unroll
      for (int kk = 0; kk < 2; ++kk) {
        v16bf a  = load_a_frag(atS, ti * 16, Nn, kk * 32, lane);
        v16bf bb = load_bt_frag(htS, td * 16, Nn, kk * 32, lane);
        acc = wmma_bf16(a, bb, acc);
      }
      const int hh = lane >> 4, d = td * 16 + (lane & 15);
      #pragma unroll
      for (int r = 0; r < 8; ++r) sS[(ti * 16 + hh * 8 + r) * Dd + d] = acc[r];
    }
    __syncthreads();

    // ---- rank-1 folded alpha/beta path (all tiny, f32 VALU) ----
    if (tid < Dd) {            // kvec = k1^T star, q2vec = q2^T star (old star)
      float kv = 0.f, qv = 0.f;
      for (int d2 = 0; d2 < Dd; ++d2) {
        float sv = stS[d2];
        kv += sv * k1[d2 * Dd + tid];
        qv += sv * q2[d2 * Dd + tid];
      }
      kvS[tid] = kv; qvS[tid] = qv;
    }
    __syncthreads();
    if (tid < Dd) {            // w = q1 @ kvec, u = k2 @ q2vec (scaled)
      float w = 0.f, u = 0.f;
      for (int j2 = 0; j2 < Dd; ++j2) {
        w += q1[tid * Dd + j2] * kvS[j2];
        u += k2[tid * Dd + j2] * qvS[j2];
      }
      wS[tid] = w * INV_SQRT_D;
      uS[tid] = u * INV_SQRT_D;
    }
    __syncthreads();
    if (tid < Nn) {            // alpha_i = sate_i . w
      float a = 0.f;
      for (int d2 = 0; d2 < Dd; ++d2) a += sS[tid * Dd + d2] * wS[d2];
      alS[tid] = a;
    }
    __syncthreads();
    for (int idx = tid; idx < Nn * Dd; idx += 256) {   // blend with star
      const int i = idx >> 7, d = idx & 127;
      const float al = alS[i];
      sS[idx] = (1.0f - al) * sS[idx] + al * stS[d];
    }
    __syncthreads();
    if (tid < Nn) {            // beta_j = sate_j . u, masked
      float bb = 0.f;
      for (int d2 = 0; d2 < Dd; ++d2) bb += sS[tid * Dd + d2] * uS[d2];
      beS[tid] = (mkS[tid] == 0.0f) ? -INFINITY : bb;
    }
    __syncthreads();
    if (wid == 0) {            // softmax over 64 nodes in one wave
      float b0 = beS[lane], b1 = beS[32 + lane];
      float m  = wave_max(fmaxf(b0, b1));
      float e0 = __expf(b0 - m), e1 = __expf(b1 - m);
      float inv = 1.0f / wave_sum(e0 + e1);
      beS[lane] = e0 * inv; beS[32 + lane] = e1 * inv;
    }
    __syncthreads();
    if (tid < Dd) {            // star = beta @ sate
      float s = 0.f;
      for (int j2 = 0; j2 < Nn; ++j2) s += beS[j2] * sS[j2 * Dd + tid];
      stS[tid] = s;
    }
    for (int idx = tid; idx < Nn * Dd; idx += 256) {   // refresh bf16 copies
      const int n = idx >> 7, d = idx & 127;
      bf16_t bv = (bf16_t)sS[idx];
      hsS[idx] = bv; htS[d * Nn + n] = bv;
    }
    __syncthreads();
  }

  // ---- gate = sigmoid([hidden, sate] @ W_lin^T)  (WMMA, K=256) ----
  float* Og = out + (size_t)b * Nn * Dd;
  for (int t = wid; t < 32; t += 8) {
    const int ti = t >> 3, td = t & 7;
    v8f acc = {0,0,0,0,0,0,0,0};
    #pragma unroll
    for (int kk = 0; kk < 8; ++kk) {
      const int kb = kk * 32;
      v16bf a  = (kb < Dd) ? load_a_frag(hbS, ti * 16, Dd, kb, lane)
                           : load_a_frag(hsS, ti * 16, Dd, kb - Dd, lane);
      v16bf bb = load_bt_frag(wl, td * 16, 2 * Dd, kb, lane);  // W_lin rows == B^T
      acc = wmma_bf16(a, bb, acc);
    }
    const int hh = lane >> 4, d = td * 16 + (lane & 15);
    #pragma unroll
    for (int r = 0; r < 8; ++r) {
      const int n = ti * 16 + hh * 8 + r;
      const float g = 1.0f / (1.0f + __expf(-acc[r]));
      Og[n * Dd + d] = g * Hg[n * Dd + d] + (1.0f - g) * sS[n * Dd + d];
    }
  }
  if (tid < Dd) star_out[(size_t)b * Dd + tid] = stS[tid];
}

extern "C" void kernel_launch(void* const* d_in, const int* in_sizes, int n_in,
                              void* d_out, int out_size, void* d_ws, size_t ws_size,
                              hipStream_t stream) {
  (void)in_sizes; (void)n_in; (void)out_size; (void)ws_size;
  const int B = 2048, N = 64, D = 128;
  const float* hidden = (const float*)d_in[0];
  const int*   adj    = (const int*)  d_in[1];
  const float* mask   = (const float*)d_in[2];
  const float* a0     = (const float*)d_in[3];
  const float* a1     = (const float*)d_in[4];
  const float* a2     = (const float*)d_in[5];
  const float* a3     = (const float*)d_in[6];
  const float* q1     = (const float*)d_in[7];
  const float* k1     = (const float*)d_in[8];
  const float* q2     = (const float*)d_in[9];
  const float* k2     = (const float*)d_in[10];
  const float* wlin   = (const float*)d_in[11];

  bf16_t* wl_bf = (bf16_t*)d_ws;                 // 128*256 bf16 = 64 KB
  const int nwl = D * 2 * D;
  wlin_to_bf16<<<(nwl + 255) / 256, 256, 0, stream>>>(wlin, wl_bf, nwl);

  float* out      = (float*)d_out;
  float* star_out = out + (size_t)B * N * D;
  star_agg_kernel<<<B, 256, 0, stream>>>(hidden, adj, mask, a0, a1, a2, a3,
                                         q1, k1, q2, k2, wl_bf, out, star_out);
}
